// PositionalCrossAttention_48644799594819
// MI455X (gfx1250) — matrix operand order repaired and checked by reading, not compiled, not run on hardware
//
#include <hip/hip_runtime.h>
#include <hip/hip_bf16.h>

// ---------------------------------------------------------------------------
// PositionalCrossAttention fused CDNA5 (gfx1250) implementation.
// B=64, P=256, C_in=4, C_out=8, H=512, heads=8, d=64.
// ---------------------------------------------------------------------------

typedef __attribute__((ext_vector_type(16))) __bf16 v16bf;
typedef __attribute__((ext_vector_type(8)))  __bf16 v8bf;
typedef __attribute__((ext_vector_type(8)))  float  v8f;

__device__ __forceinline__ unsigned short f2bf(float f) {
    unsigned int u = __float_as_uint(f);
    unsigned int r = u + 0x7fffu + ((u >> 16) & 1u);   // round-to-nearest-even
    return (unsigned short)(r >> 16);
}

// ---------------------------------------------------------------------------
// Kernel 1: positional-encoding MLP (+ q projection for target rows).
// 768 rows total: rows [0,256) = input positions -> ipe; rows [256,768) =
// target positions -> pe -> q (pe itself not needed downstream).
// 4 rows per block so pe_w2 / wq rows are each read once per 4 rows.
// ---------------------------------------------------------------------------
__global__ void pca_pe_kernel(const float* __restrict__ ipos,
                              const float* __restrict__ tpos,
                              const float* __restrict__ w1, const float* __restrict__ b1,
                              const float* __restrict__ w2, const float* __restrict__ b2,
                              const float* __restrict__ wq, const float* __restrict__ bq,
                              float* __restrict__ ipe, float* __restrict__ qout) {
    __shared__ float sH[4][512];
    __shared__ float sP[4][512];
    const int t    = threadIdx.x;       // 256 threads
    const int row0 = blockIdx.x * 4;    // 192 blocks
    const bool isTarget = (row0 >= 256);

    float px[4], py[4], pz[4];
    for (int r = 0; r < 4; ++r) {
        int row = row0 + r;
        const float* p = (row < 256) ? (ipos + row * 3) : (tpos + (row - 256) * 3);
        px[r] = p[0]; py[r] = p[1]; pz[r] = p[2];
    }

    // Layer 1: 3 -> 512, ReLU
    for (int rr = 0; rr < 2; ++rr) {
        int o = t + rr * 256;
        float wa = w1[o * 3], wb = w1[o * 3 + 1], wc = w1[o * 3 + 2], bb = b1[o];
        for (int r = 0; r < 4; ++r) {
            float h = px[r] * wa + py[r] * wb + pz[r] * wc + bb;
            sH[r][o] = h > 0.f ? h : 0.f;
        }
    }
    __syncthreads();

    // Layer 2: 512 -> 512
    for (int rr = 0; rr < 2; ++rr) {
        int o = t + rr * 256;
        const float* wrow = w2 + (long)o * 512;
        float a0 = 0.f, a1 = 0.f, a2 = 0.f, a3 = 0.f;
        for (int i = 0; i < 512; ++i) {
            float w = wrow[i];
            a0 += sH[0][i] * w; a1 += sH[1][i] * w;
            a2 += sH[2][i] * w; a3 += sH[3][i] * w;
        }
        float bb = b2[o];
        if (!isTarget) {
            ipe[(row0 + 0) * 512 + o] = a0 + bb;
            ipe[(row0 + 1) * 512 + o] = a1 + bb;
            ipe[(row0 + 2) * 512 + o] = a2 + bb;
            ipe[(row0 + 3) * 512 + o] = a3 + bb;
        } else {
            sP[0][o] = a0 + bb; sP[1][o] = a1 + bb;
            sP[2][o] = a2 + bb; sP[3][o] = a3 + bb;
        }
    }

    if (isTarget) {
        __syncthreads();
        // q = pe @ wq^T + bq
        for (int rr = 0; rr < 2; ++rr) {
            int o = t + rr * 256;
            const float* wrow = wq + (long)o * 512;
            float a0 = 0.f, a1 = 0.f, a2 = 0.f, a3 = 0.f;
            for (int i = 0; i < 512; ++i) {
                float w = wrow[i];
                a0 += sP[0][i] * w; a1 += sP[1][i] * w;
                a2 += sP[2][i] * w; a3 += sP[3][i] * w;
            }
            float bb = bq[o];
            int qr = row0 - 256;
            qout[(qr + 0) * 512 + o] = a0 + bb;
            qout[(qr + 1) * 512 + o] = a1 + bb;
            qout[(qr + 2) * 512 + o] = a2 + bb;
            qout[(qr + 3) * 512 + o] = a3 + bb;
        }
    }
}

// ---------------------------------------------------------------------------
// Kernel 2: pack wk / wv / wo (f32, [out=512][in=512] row-major) into bf16
// WMMA B-panels: layout [kc(16)][nt(32)][lane(32)][elem(16)], where for
// B(32x16) per CDNA5 ISA: lane L holds column N = L&15, K =
// kc*32 + (L>=16 ? 8 : 0) + (elem&7) + (elem>=8 ? 16 : 0).  One aligned 32B load per lane per WMMA.
// ---------------------------------------------------------------------------
__global__ void pca_pack_kernel(const float* __restrict__ wk,
                                const float* __restrict__ wv,
                                const float* __restrict__ wo,
                                unsigned short* __restrict__ bpk) {
    int gid = blockIdx.x * 256 + threadIdx.x;   // < 3*262144
    int mat = gid >> 18;
    int idx = gid & 262143;
    int j    = idx & 15;
    int lane = (idx >> 4) & 31;
    int nt   = (idx >> 9) & 31;
    int kc   = idx >> 14;                        // 0..15
    int o = nt * 16 + (lane & 15);
    int k = kc * 32 + ((lane >> 4) << 3) + (j & 7) + ((j >> 3) << 4);
    const float* w = (mat == 0) ? wk : ((mat == 1) ? wv : wo);
    bpk[gid] = f2bf(w[(long)o * 512 + k]);
}

// ---------------------------------------------------------------------------
// Kernel 3: fused main kernel.  One block = (batch b, tile of 8 patches).
//   Phase 1: patch = emb + ipe  -> bf16 in LDS (32 rows x 512)
//   Phase 2: K,V = patch @ w^T + b   via v_wmma_f32_16x16x32_bf16 -> f32 LDS
//   Phase 3: per-(p,h) 8x4 softmax attention in f32 VALU; ctx -> bf16 LDS
//   Phase 4: out = ctx @ wo^T + bo   via WMMA -> HBM (f32)
// LDS: [0,64K): phase1 patch-bf16 (32KB) overlaid later by ctx-bf16 (64KB);
//      [64K,128K): K f32; [128K,192K): V f32.  Total 192 KB (<=320 KB/WGP).
// ---------------------------------------------------------------------------
__global__ void pca_main_kernel(const float* __restrict__ emb,
                                const float* __restrict__ ipe,
                                const float* __restrict__ q,
                                const unsigned short* __restrict__ bpk,
                                const float* __restrict__ bk,
                                const float* __restrict__ bv,
                                const float* __restrict__ bo,
                                float* __restrict__ out) {
    extern __shared__ char smem[];
    unsigned short* sX = (unsigned short*)smem;                  // 32x512 bf16
    unsigned short* sC = (unsigned short*)smem;                  // 64x512 bf16 (reuses sX region)
    float* sK = (float*)(smem + 65536);                          // 32x512 f32
    float* sV = (float*)(smem + 131072);                         // 32x512 f32

    const int bid  = blockIdx.x;           // 2048 blocks
    const int b    = bid >> 5;
    const int p0   = (bid & 31) * 8;
    const int tid  = threadIdx.x;
    const int w    = tid >> 5;             // wave 0..7
    const int lane = tid & 31;

    // ---- Phase 1: load patch rows, add ipe, convert to bf16 -> sX ----
    for (int i = 0; i < 16; ++i) {
        int s   = tid + i * 256;           // 4096 float4 slots
        int row = s >> 7;                  // 0..31 (= p_local*4 + c)
        int c4  = (s & 127) * 4;
        int pl  = row >> 2, c = row & 3;
        const float4 e  = *(const float4*)(emb + (((long)(b * 256 + p0 + pl) * 4 + c) * 512 + c4));
        const float4 pp = *(const float4*)(ipe + ((b * 4 + c) * 512 + c4));
        unsigned short* d = sX + row * 512 + c4;
        d[0] = f2bf(e.x + pp.x); d[1] = f2bf(e.y + pp.y);
        d[2] = f2bf(e.z + pp.z); d[3] = f2bf(e.w + pp.w);
    }
    __syncthreads();

    // ---- Phase 2: K and V projections (WMMA) ----
    {
        const int mt = w & 1;              // M-tile (patch rows 0-15 / 16-31)
        const int ng = w >> 1;             // N-tile group: nt = ng*8 .. ng*8+7
        for (int mat = 0; mat < 2; ++mat) {
            const unsigned short* wb = bpk + mat * 262144;
            const float* bias = (mat == 0) ? bk : bv;
            float* dst = (mat == 0) ? sK : sV;
            v8f acc[8];
            for (int i = 0; i < 8; ++i)
                for (int e = 0; e < 8; ++e) acc[i][e] = 0.f;
            for (int kc = 0; kc < 16; ++kc) {
                // A-matrix chunk per CDNA5 ISA 16-bit 16x32 layout
                int arow = mt * 16 + (lane & 15);
                int kb   = kc * 32 + ((lane >> 4) << 3);
                v8bf alo = *(const v8bf*)(sX + arow * 512 + kb);
                v8bf ahi = *(const v8bf*)(sX + arow * 512 + kb + 16);
                v16bf a;
                #pragma unroll
                for (int e = 0; e < 8; ++e) { a[e] = alo[e]; a[e + 8] = ahi[e]; }
                #pragma unroll
                for (int i = 0; i < 8; ++i) {
                    int nt = ng * 8 + i;
                    v16bf bm = *(const v16bf*)(wb + (((kc * 32 + nt) * 32 + lane) << 4));
                    acc[i] = __builtin_amdgcn_wmma_f32_16x16x32_bf16(
                        false, a, false, bm, (short)0, acc[i], false, false);
                }
            }
            // bias add + scatter to LDS (C/D layout: m = r + (lane>=16?8:0), n = lane&15)
            const int mbase = mt * 16 + ((lane >> 4) << 3);
            for (int i = 0; i < 8; ++i) {
                int col  = (ng * 8 + i) * 16 + (lane & 15);
                float bb = bias[col];
                #pragma unroll
                for (int r = 0; r < 8; ++r)
                    dst[(mbase + r) * 512 + col] = acc[i][r] + bb;
            }
        }
    }
    __syncthreads();

    // ---- Phase 3: attention (f32 VALU), ctx -> bf16 in sC ----
    {
        const int pair = tid >> 2;         // 0..63 : (p_local, head)
        const int pl   = pair >> 3;
        const int h    = pair & 7;
        const int j    = tid & 3;
        const float* qb = q + (long)(b * 8) * 512 + h * 64;
        const float* v0r = sV + (pl * 4 + 0) * 512 + h * 64;
        const float* v1r = sV + (pl * 4 + 1) * 512 + h * 64;
        const float* v2r = sV + (pl * 4 + 2) * 512 + h * 64;
        const float* v3r = sV + (pl * 4 + 3) * 512 + h * 64;
        for (int qi = 0; qi < 2; ++qi) {
            int qq = j + qi * 4;
            const float* qv = qb + qq * 512;
            float l[4];
            for (int c = 0; c < 4; ++c) {
                const float* kr = sK + (pl * 4 + c) * 512 + h * 64;
                float s = 0.f;
                for (int d = 0; d < 64; ++d) s += qv[d] * kr[d];
                l[c] = s * 0.125f;         // 1/sqrt(64)
            }
            float m  = fmaxf(fmaxf(l[0], l[1]), fmaxf(l[2], l[3]));
            float e0 = __expf(l[0] - m), e1 = __expf(l[1] - m);
            float e2 = __expf(l[2] - m), e3 = __expf(l[3] - m);
            float inv = 1.f / (e0 + e1 + e2 + e3);
            float a0 = e0 * inv, a1 = e1 * inv, a2 = e2 * inv, a3 = e3 * inv;
            unsigned short* crow = sC + (pl * 8 + qq) * 512 + h * 64;
            for (int d = 0; d < 64; ++d)
                crow[d] = f2bf(a0 * v0r[d] + a1 * v1r[d] + a2 * v2r[d] + a3 * v3r[d]);
        }
    }
    __syncthreads();

    // ---- Phase 4: output projection (WMMA) -> HBM ----
    {
        const unsigned short* wb = bpk + 2 * 262144;
        for (int rep = 0; rep < 2; ++rep) {
            const int mt = (w >> 2) * 2 + rep;   // 0..3 (ctx rows 64 = 4 M-tiles)
            const int ng = w & 3;                // nt = ng*8 .. ng*8+7
            v8f acc[8];
            for (int i = 0; i < 8; ++i)
                for (int e = 0; e < 8; ++e) acc[i][e] = 0.f;
            for (int kc = 0; kc < 16; ++kc) {
                int arow = mt * 16 + (lane & 15);
                int kb   = kc * 32 + ((lane >> 4) << 3);
                v8bf alo = *(const v8bf*)(sC + arow * 512 + kb);
                v8bf ahi = *(const v8bf*)(sC + arow * 512 + kb + 16);
                v16bf a;
                #pragma unroll
                for (int e = 0; e < 8; ++e) { a[e] = alo[e]; a[e + 8] = ahi[e]; }
                #pragma unroll
                for (int i = 0; i < 8; ++i) {
                    int nt = ng * 8 + i;
                    v16bf bm = *(const v16bf*)(wb + (((kc * 32 + nt) * 32 + lane) << 4));
                    acc[i] = __builtin_amdgcn_wmma_f32_16x16x32_bf16(
                        false, a, false, bm, (short)0, acc[i], false, false);
                }
            }
            const int mbase = mt * 16 + ((lane >> 4) << 3);
            for (int i = 0; i < 8; ++i) {
                int col  = (ng * 8 + i) * 16 + (lane & 15);
                float bb = bo[col];
                #pragma unroll
                for (int r = 0; r < 8; ++r) {
                    int row = mbase + r;            // 0..63 = p_local*8 + qq
                    int pl = row >> 3, qq = row & 7;
                    out[((long)((b * 256 + p0 + pl) * 8 + qq)) * 512 + col] = acc[i][r] + bb;
                }
            }
        }
    }
}

// ---------------------------------------------------------------------------
// Host launcher
// ---------------------------------------------------------------------------
extern "C" void kernel_launch(void* const* d_in, const int* in_sizes, int n_in,
                              void* d_out, int out_size, void* d_ws, size_t ws_size,
                              hipStream_t stream) {
    const float* emb  = (const float*)d_in[0];
    const float* ipos = (const float*)d_in[1];
    const float* tpos = (const float*)d_in[2];
    const float* pw1  = (const float*)d_in[3];
    const float* pb1  = (const float*)d_in[4];
    const float* pw2  = (const float*)d_in[5];
    const float* pb2  = (const float*)d_in[6];
    const float* wq   = (const float*)d_in[7];
    const float* bq   = (const float*)d_in[8];
    const float* wk   = (const float*)d_in[9];
    const float* bk   = (const float*)d_in[10];
    const float* wv   = (const float*)d_in[11];
    const float* bv   = (const float*)d_in[12];
    const float* wo   = (const float*)d_in[13];
    const float* bo   = (const float*)d_in[14];
    float* out = (float*)d_out;

    char* ws = (char*)d_ws;
    float*          ipe  = (float*)ws;                          // 512 KB
    float*          qbuf = (float*)(ws + 524288);               // 1 MB
    unsigned short* bpk  = (unsigned short*)(ws + 1572864);     // 1.5 MB bf16 panels

    pca_pe_kernel<<<192, 256, 0, stream>>>(ipos, tpos, pw1, pb1, pw2, pb2, wq, bq, ipe, qbuf);
    pca_pack_kernel<<<3072, 256, 0, stream>>>(wk, wv, wo, bpk);
    pca_main_kernel<<<2048, 256, 196608, stream>>>(emb, ipe, qbuf, bpk, bk, bv, bo, out);
}
